// SetCriterion_32409823215669
// MI455X (gfx1250) — matrix operand order repaired and checked by reading, not compiled, not run on hardware
//
#include <hip/hip_runtime.h>
#include <hip/hip_bf16.h>

#define N_OBJ    40960
#define DFEAT    256
#define P_EDGES  200000
#define TAKE_NEG 100000
#define HIDDEN   512
#define KDIM     512           // tok_i(256) + tok_j(256); relation part folded into rel_contrib
#define NPOW     262144        // 2^18 >= P_EDGES, for bitonic sort
#define TILE_M   64            // edges per block
#define NWAVES   16            // 512-thread blocks

typedef __attribute__((ext_vector_type(16))) __bf16 v16bf;
typedef __attribute__((ext_vector_type(8)))  float  v8f;

union V16 { v16bf v; uint4 u[2]; };

__device__ __forceinline__ unsigned short f2bf(float f) {
    unsigned u = __float_as_uint(f);
    unsigned r = u + 0x7FFFu + ((u >> 16) & 1u);   // round-to-nearest-even
    return (unsigned short)(r >> 16);
}

// ---------------------------------------------------------------------------
// rel_contrib[n] = b1[n] + sum_k relation_token[k] * W1[512+k][n]   (512 outputs)
// ---------------------------------------------------------------------------
__global__ void relc_kernel(const float* __restrict__ rel,
                            const float* __restrict__ W1,
                            const float* __restrict__ b1,
                            float* __restrict__ relc) {
    int n = blockIdx.x * 256 + threadIdx.x;
    if (n >= HIDDEN) return;
    float acc = b1[n];
    #pragma unroll 4
    for (int k = 0; k < DFEAT; ++k)
        acc += rel[k] * W1[(size_t)(KDIM + k) * HIDDEN + n];
    relc[n] = acc;
}

// ---------------------------------------------------------------------------
// Pre-swizzle W1[0:512][0:512] fp32 -> bf16 WMMA B-fragments.
//   lane<16 : column n = lane,     K = 0..15
//   lane>=16: column n = lane-16,  K = 16..31
// frag[kt][nt][lane][16 halves], kt in [0,16), nt in [0,32)
// ---------------------------------------------------------------------------
__global__ void swizzle_w1_kernel(const float* __restrict__ W1,
                                  unsigned short* __restrict__ frag) {
    unsigned gid = blockIdx.x * 256 + threadIdx.x;       // 262144 total
    unsigned h    =  gid        & 15u;
    unsigned lane = (gid >> 4)  & 31u;
    unsigned nt   = (gid >> 9)  & 31u;
    unsigned kt   =  gid >> 14;
    unsigned n = nt * 16 + (lane & 15u);
    unsigned k = kt * 32 + (lane >> 4) * 16 + h;
    frag[gid] = f2bf(W1[(size_t)k * HIDDEN + n]);
}

// ---------------------------------------------------------------------------
// Pad sort keys beyond P with max value (sorts last in ascending order)
// ---------------------------------------------------------------------------
__global__ void pad_keys_kernel(unsigned long long* __restrict__ keys) {
    int i = P_EDGES + blockIdx.x * 256 + threadIdx.x;
    if (i < NPOW) keys[i] = ~0ull;
}

// ---------------------------------------------------------------------------
// Fused: gather -> bf16 WMMA GEMM (64 edges x 512 hidden, K=512) -> ReLU
//        -> 512x2 GEMM -> softmax -> score + sort key.
// Block: 512 threads (16 waves). Wave w owns hidden columns [32w, 32w+32).
// Each wave: 4 M-tiles x 2 N-tiles; each B fragment per K-step feeds 4 WMMAs,
// while accumulators stay at 64 VGPRs/lane (no VGPR-MSB addressing needed).
// ---------------------------------------------------------------------------
__global__ __launch_bounds__(512) void gemm_score_kernel(
        const float* __restrict__ tok,          // N_OBJ x 256
        const int*   __restrict__ edges,        // P x 2
        const unsigned short* __restrict__ w1frag,
        const float* __restrict__ relc,         // 512 (includes b1)
        const float* __restrict__ W2,           // 512 x 2
        const float* __restrict__ b2,           // 2
        float* __restrict__ scores_out,         // P
        unsigned long long* __restrict__ keys)  // NPOW
{
    __shared__ unsigned short sA[TILE_M][KDIM + 8]; // 64 edges x 512 bf16, padded
    __shared__ int   s_idx[2 * TILE_M];
    __shared__ float s_part[NWAVES][TILE_M][2];     // per-wave logit partials

    const int t    = threadIdx.x;
    const int w    = t >> 5;
    const int lane = t & 31;
    const int ebase = blockIdx.x * TILE_M;

    if (t < 2 * TILE_M) s_idx[t] = edges[(ebase + (t >> 1)) * 2 + (t & 1)];
    __syncthreads();

    // ---- stage A tile (gathered tokens) into LDS as bf16: 128 half-rows ----
    #pragma unroll
    for (int r = 0; r < 8; ++r) {
        int hrow = w * 8 + r;            // 0..127
        int m = hrow >> 1, tk = hrow & 1;
        int row = s_idx[m * 2 + tk];
        const float4* src = (const float4*)(tok + (size_t)row * DFEAT);
        float4 a = src[lane * 2];
        float4 b = src[lane * 2 + 1];
        uint4 p;
        p.x = (unsigned)f2bf(a.x) | ((unsigned)f2bf(a.y) << 16);
        p.y = (unsigned)f2bf(a.z) | ((unsigned)f2bf(a.w) << 16);
        p.z = (unsigned)f2bf(b.x) | ((unsigned)f2bf(b.y) << 16);
        p.w = (unsigned)f2bf(b.z) | ((unsigned)f2bf(b.w) << 16);
        *(uint4*)&sA[m][tk * DFEAT + lane * 8] = p;
    }
    __syncthreads();

    // ---- WMMA main loop: 4 M-tiles x 2 N-tiles per wave, 16 K-steps ----
    v8f acc[4][2] = {};
    const unsigned short* aRow[4];
    #pragma unroll
    for (int mt = 0; mt < 4; ++mt) aRow[mt] = &sA[mt * 16 + (lane & 15)][0];
    const int hs = (lane >> 4) * 16;     // A-fragment K-half select
    const uint4* bp = (const uint4*)w1frag;
    // fragment (kt, nt=w*2+i) at uint4 index: ((kt*32 + nt)*32 + lane)*2
    const int bbase = ((w * 2) * 32 + lane) * 2;   // kt stride 2048, i stride 64

    for (int kt = 0; kt < 16; ++kt) {
        const int k0 = kt * 32;
        V16 a[4], b[2];
        #pragma unroll
        for (int mt = 0; mt < 4; ++mt) {
            a[mt].u[0] = *(const uint4*)(aRow[mt] + k0 + hs);
            a[mt].u[1] = *(const uint4*)(aRow[mt] + k0 + hs + 8);
        }
        const int bofs = bbase + kt * 2048;
        #pragma unroll
        for (int i = 0; i < 2; ++i) {
            b[i].u[0] = bp[bofs + i * 64];
            b[i].u[1] = bp[bofs + i * 64 + 1];
        }
        #pragma unroll
        for (int i = 0; i < 2; ++i) {
            #pragma unroll
            for (int mt = 0; mt < 4; ++mt)
                acc[mt][i] = __builtin_amdgcn_wmma_f32_16x16x32_bf16(
                    false, a[mt].v, false, b[i].v, (short)0, acc[mt][i], false, false);
        }
    }

    // ---- epilogue: +rel_contrib, ReLU, x W2 (512x2) partials ----
    float rc[2], w2a[2], w2b[2];
    #pragma unroll
    for (int i = 0; i < 2; ++i) {
        const int n = (w * 2 + i) * 16 + (lane & 15);
        rc[i]  = relc[n];
        w2a[i] = W2[n * 2];
        w2b[i] = W2[n * 2 + 1];
    }
    const int mb = (lane >> 4) * 8;      // lanes 0..15 -> m=r, 16..31 -> m=8+r
    #pragma unroll
    for (int mt = 0; mt < 4; ++mt) {
        float lp0[8], lp1[8];
        #pragma unroll
        for (int r = 0; r < 8; ++r) { lp0[r] = 0.f; lp1[r] = 0.f; }
        #pragma unroll
        for (int i = 0; i < 2; ++i) {
            #pragma unroll
            for (int r = 0; r < 8; ++r) {
                float hv = acc[mt][i][r] + rc[i];
                hv = fmaxf(hv, 0.f);
                lp0[r] += hv * w2a[i];
                lp1[r] += hv * w2b[i];
            }
        }
        // reduce across 16 lanes (same m, different n); halves stay independent
        #pragma unroll
        for (int off = 1; off < 16; off <<= 1) {
            #pragma unroll
            for (int r = 0; r < 8; ++r) {
                lp0[r] += __shfl_xor(lp0[r], off, 32);
                lp1[r] += __shfl_xor(lp1[r], off, 32);
            }
        }
        if ((lane & 15) == 0) {
            #pragma unroll
            for (int r = 0; r < 8; ++r) {
                s_part[w][mt * 16 + mb + r][0] = lp0[r];
                s_part[w][mt * 16 + mb + r][1] = lp1[r];
            }
        }
    }
    __syncthreads();

    if (t < TILE_M) {
        float l0 = b2[0], l1 = b2[1];
        #pragma unroll
        for (int ww = 0; ww < NWAVES; ++ww) {
            l0 += s_part[ww][t][0];
            l1 += s_part[ww][t][1];
        }
        const float score = 1.f / (1.f + __expf(l0 - l1)); // softmax[:,1]
        const int e = ebase + t;
        scores_out[e] = score;
        unsigned u = __float_as_uint(score);
        unsigned s = (u & 0x80000000u) ? ~u : (u | 0x80000000u); // monotonic map
        // descending score, ascending index tiebreak, ascending u64 sort:
        keys[e] = ((unsigned long long)(~s) << 32) | (unsigned)e;
    }
}

// ---------------------------------------------------------------------------
// Global-memory bitonic sort step (ascending overall)
// ---------------------------------------------------------------------------
__global__ void bitonic_kernel(unsigned long long* __restrict__ keys, int j, int k) {
    int i = blockIdx.x * 256 + threadIdx.x;
    int l = i ^ j;
    if (l > i) {
        unsigned long long a = keys[i];
        unsigned long long b = keys[l];
        bool asc = ((i & k) == 0);
        if ((a > b) == asc) { keys[i] = b; keys[l] = a; }
    }
}

// ---------------------------------------------------------------------------
// Gather top-K edges in sorted order
// ---------------------------------------------------------------------------
__global__ void select_kernel(const unsigned long long* __restrict__ keys,
                              const int* __restrict__ edges,
                              float* __restrict__ out_sel) {
    int r = blockIdx.x * 256 + threadIdx.x;
    if (r < TAKE_NEG) {
        unsigned e = (unsigned)(keys[r] & 0xFFFFFFFFu);
        out_sel[2 * r]     = (float)edges[e * 2];
        out_sel[2 * r + 1] = (float)edges[e * 2 + 1];
    }
}

extern "C" void kernel_launch(void* const* d_in, const int* in_sizes, int n_in,
                              void* d_out, int out_size, void* d_ws, size_t ws_size,
                              hipStream_t stream) {
    const float* tok  = (const float*)d_in[0];   // 40960 x 256
    const float* rel  = (const float*)d_in[1];   // 256
    const int*   edg  = (const int*)  d_in[2];   // 200000 x 2
    const float* W1   = (const float*)d_in[3];   // 768 x 512
    const float* b1   = (const float*)d_in[4];   // 512
    const float* W2   = (const float*)d_in[5];   // 512 x 2
    const float* b2   = (const float*)d_in[6];   // 2
    float* out = (float*)d_out;                  // [200000 selected | 200000 scores]

    char* ws = (char*)d_ws;
    unsigned long long* keys = (unsigned long long*)ws;                       // 2 MB
    unsigned short* w1frag = (unsigned short*)(ws + (size_t)NPOW * 8);        // 512 KB
    float* relc = (float*)(ws + (size_t)NPOW * 8 + (size_t)KDIM * HIDDEN * 2);// 2 KB
    float* scores = out + 2 * TAKE_NEG;

    pad_keys_kernel<<<(NPOW - P_EDGES + 255) / 256, 256, 0, stream>>>(keys);
    relc_kernel<<<2, 256, 0, stream>>>(rel, W1, b1, relc);
    swizzle_w1_kernel<<<(KDIM * HIDDEN) / 256, 256, 0, stream>>>(W1, w1frag);

    gemm_score_kernel<<<P_EDGES / TILE_M, 512, 0, stream>>>(
        tok, edg, w1frag, relc, W2, b2, scores, keys);

    for (int k = 2; k <= NPOW; k <<= 1)
        for (int j = k >> 1; j > 0; j >>= 1)
            bitonic_kernel<<<NPOW / 256, 256, 0, stream>>>(keys, j, k);

    select_kernel<<<(TAKE_NEG + 255) / 256, 256, 0, stream>>>(keys, edg, out);
}
